// GNN_62508954026571
// MI455X (gfx1250) — matrix operand order repaired and checked by reading, not compiled, not run on hardware
//
#include <hip/hip_runtime.h>
#include <hip/hip_bf16.h>

typedef __attribute__((ext_vector_type(16))) __bf16 v16bf;
typedef __attribute__((ext_vector_type(8)))  __bf16 v8bf;
typedef __attribute__((ext_vector_type(8)))  float  v8f;

#define DIM 128
#define LDS_PITCH 136   // 128 + 8 pad (272B rows: 16B aligned, bank-staggered)

__device__ __forceinline__ float4 ld4(const float* p){ return *(const float4*)p; }

#define AFF4(f,s,t) do{ \
  f.x = fmaxf(f.x*s.x + t.x, 0.f); \
  f.y = fmaxf(f.y*s.y + t.y, 0.f); \
  f.z = fmaxf(f.z*s.z + t.z, 0.f); \
  f.w = fmaxf(f.w*s.w + t.w, 0.f); }while(0)

// ---------------------------------------------------------------------------
// WMMA GEMM: Y = post( pre(X) @ W ),  X:[N,128] fp32 row-major, W:[128,128]
// pre : optional per-column affine+relu (fused BN)   (PRE=1)
// post: optional +bias, relu                         (POST=1)
// Block = 256 thr = 8 waves; block tile 128 rows x 128 cols; wave = 16 rows.
// ---------------------------------------------------------------------------
template<int PRE, int POST>
__global__ __launch_bounds__(256) void gemm128_kernel(
    const float* __restrict__ X, const float* __restrict__ W,
    const float* __restrict__ pre_scale, const float* __restrict__ pre_shift,
    const float* __restrict__ bias, float* __restrict__ Y, int N)
{
  __shared__ __bf16 wt[DIM * LDS_PITCH];   // W transposed: wt[n][k], bf16
  const int tid = threadIdx.x;
  for (int idx = tid; idx < DIM * DIM; idx += 256) {
    int k = idx >> 7, n = idx & 127;
    wt[n * LDS_PITCH + k] = (__bf16)W[idx];
  }
  __syncthreads();

  const int wave = tid >> 5;
  const int lane = tid & 31;
  const int hi   = lane >> 4;       // half-wave select
  const int lid  = lane & 15;
  const int m0   = blockIdx.x * 128 + wave * 16;
  const int mrow = m0 + lid;
  const int mload = (mrow < N) ? mrow : (N - 1);
  const float* xrow = X + (size_t)mload * DIM;

  v8f acc[8];
#pragma unroll
  for (int i = 0; i < 8; i++) acc[i] = (v8f){0,0,0,0,0,0,0,0};

#pragma unroll
  for (int kt = 0; kt < 4; kt++) {
    const int k0 = kt * 32;
    const int ka = k0 + hi * 8;     // A k-base for this half-wave
    float4 f0 = ld4(xrow + ka);
    float4 f1 = ld4(xrow + ka + 4);
    float4 f2 = ld4(xrow + ka + 16);
    float4 f3 = ld4(xrow + ka + 20);
    if (PRE) {
      float4 s0 = ld4(pre_scale + ka),      s1 = ld4(pre_scale + ka + 4);
      float4 s2 = ld4(pre_scale + ka + 16), s3 = ld4(pre_scale + ka + 20);
      float4 t0 = ld4(pre_shift + ka),      t1 = ld4(pre_shift + ka + 4);
      float4 t2 = ld4(pre_shift + ka + 16), t3 = ld4(pre_shift + ka + 20);
      AFF4(f0, s0, t0); AFF4(f1, s1, t1); AFF4(f2, s2, t2); AFF4(f3, s3, t3);
    }
    // A fragment per ISA: lane m=lid, kb=hi*8; a[e]=A[m][k0+kb+e(+8 if e>=8)]
    v16bf a;
    a[0]=(__bf16)f0.x;  a[1]=(__bf16)f0.y;  a[2]=(__bf16)f0.z;  a[3]=(__bf16)f0.w;
    a[4]=(__bf16)f1.x;  a[5]=(__bf16)f1.y;  a[6]=(__bf16)f1.z;  a[7]=(__bf16)f1.w;
    a[8]=(__bf16)f2.x;  a[9]=(__bf16)f2.y;  a[10]=(__bf16)f2.z; a[11]=(__bf16)f2.w;
    a[12]=(__bf16)f3.x; a[13]=(__bf16)f3.y; a[14]=(__bf16)f3.z; a[15]=(__bf16)f3.w;

#pragma unroll
    for (int nt = 0; nt < 8; nt++) {
      // B fragment: lane n = nt*16+lid, kb=hi*8; b[e]=W[k0+kb+e(+8 if e>=8)][n]=wt[n][k0+kb+e(+8 if e>=8)]
      const __bf16* wp = &wt[(nt * 16 + lid) * LDS_PITCH + k0 + hi * 8];
      v8bf blo = *(const v8bf*)wp;
      v8bf bh  = *(const v8bf*)(wp + 16);
      v16bf b;
#pragma unroll
      for (int i = 0; i < 8; i++) { b[i] = blo[i]; b[8 + i] = bh[i]; }
      acc[nt] = __builtin_amdgcn_wmma_f32_16x16x32_bf16(
          false, a, false, b, (short)0, acc[nt], false, false);
    }
  }

  // D layout: lane n = nt*16+lid, VGPR r -> row m0 + r + 8*hi
#pragma unroll
  for (int nt = 0; nt < 8; nt++) {
    const int col = nt * 16 + lid;
    float bv = 0.f;
    if (POST) bv = bias[col];
#pragma unroll
    for (int r = 0; r < 8; r++) {
      const int row = m0 + r + hi * 8;
      if (row < N) {
        float v = acc[nt][r];
        if (POST) v = fmaxf(v + bv, 0.f);
        Y[(size_t)row * DIM + col] = v;
      }
    }
  }
}

// ---------------------------------------------------------------------------
__global__ void fill0_kernel(float* __restrict__ p, int n) {
  int i = blockIdx.x * blockDim.x + threadIdx.x;
  if (i < n) p[i] = 0.f;
}

__global__ void embed_kernel(const int* __restrict__ d, const float* __restrict__ embed_d,
                             float* __restrict__ X, int N) {
  int i = blockIdx.x * blockDim.x + threadIdx.x;
  if (i >= N * DIM) return;
  int n = i >> 7, c = i & 127;
  int dv = d[n]; dv = dv < 0 ? 0 : (dv > 5 ? 5 : dv);
  X[i] = embed_d[dv * DIM + c];
}

__global__ void counts_kernel(const int* __restrict__ bid, float* __restrict__ counts, int N) {
  int i = blockIdx.x * blockDim.x + threadIdx.x;
  if (i < N) atomicAdd(&counts[bid[i]], 1.0f);
}

__global__ void scale_kernel(const float* __restrict__ X, const float* __restrict__ eps, int l,
                             float* __restrict__ Z, int total) {
  int i = blockIdx.x * blockDim.x + threadIdx.x;
  if (i >= total) return;
  Z[i] = X[i] * (1.0f + eps[l]);
}

// one wave per edge; lane handles 4 contiguous feature columns
__global__ __launch_bounds__(256) void edge_kernel(
    const int* __restrict__ src, const int* __restrict__ dst,
    const float* __restrict__ Y, float* __restrict__ Z, int E) {
  int t = blockIdx.x * 256 + threadIdx.x;
  int e = t >> 5;
  if (e >= E) return;
  int c0 = (t & 31) * 4;
  const float4 v = *(const float4*)(Y + (size_t)src[e] * DIM + c0);
  float* zp = Z + (size_t)dst[e] * DIM + c0;
  atomicAdd(zp + 0, v.x); atomicAdd(zp + 1, v.y);
  atomicAdd(zp + 2, v.z); atomicAdd(zp + 3, v.w);
}

// per-column sum / sumsq over N rows (for training-mode BatchNorm)
__global__ __launch_bounds__(256) void colstats_kernel(
    const float* __restrict__ X, float* __restrict__ sum, float* __restrict__ sumsq, int N) {
  __shared__ float ssum[256], ssq[256];
  const int c = threadIdx.x & 127;
  const int half = threadIdx.x >> 7;
  float s = 0.f, q = 0.f;
  for (int row = blockIdx.x * 2 + half; row < N; row += gridDim.x * 2) {
    float v = X[(size_t)row * DIM + c];
    s += v; q += v * v;
  }
  ssum[threadIdx.x] = s; ssq[threadIdx.x] = q;
  __syncthreads();
  if (half == 0) {
    atomicAdd(&sum[c],   ssum[c] + ssum[c + 128]);
    atomicAdd(&sumsq[c], ssq[c]  + ssq[c + 128]);
  }
}

__global__ void bnfin_kernel(const float* __restrict__ sum, const float* __restrict__ sumsq,
                             const float* __restrict__ g, const float* __restrict__ b,
                             float invN, float* __restrict__ scale, float* __restrict__ shift) {
  int c = threadIdx.x;   // 128 threads
  float mu  = sum[c] * invN;
  float var = sumsq[c] * invN - mu * mu;
  float inv = rsqrtf(var + 1e-5f);
  float sc  = inv * g[c];
  scale[c] = sc;
  shift[c] = b[c] - mu * sc;
}

template<int ACCUM>
__global__ void bnrelu_kernel(const float* __restrict__ X, const float* __restrict__ scale,
                              const float* __restrict__ shift, float* __restrict__ Y,
                              const int* __restrict__ bid, float* __restrict__ hsum, int N) {
  int i = blockIdx.x * blockDim.x + threadIdx.x;
  if (i >= N * DIM) return;
  int n = i >> 7, c = i & 127;
  float v = fmaxf(X[i] * scale[c] + shift[c], 0.f);
  Y[i] = v;
  if (ACCUM) atomicAdd(&hsum[(size_t)bid[n] * DIM + c], v);
}

__global__ void segfin_kernel(const float* __restrict__ hsum, const float* __restrict__ counts,
                              float* __restrict__ out, int total) {
  int i = blockIdx.x * blockDim.x + threadIdx.x;
  if (i >= total) return;
  out[i] = hsum[i] / fmaxf(counts[i >> 7], 1.0f);
}

// predict MLP: [G,128] -> relu(@W1[128,256]+b1) -> @W2[256,1]+b2
__global__ __launch_bounds__(128) void pred_kernel(
    const float* __restrict__ hg, const float* __restrict__ W1, const float* __restrict__ b1,
    const float* __restrict__ W2, const float* __restrict__ b2, float* __restrict__ out) {
  __shared__ float h[DIM];
  __shared__ float red[DIM];
  const int g = blockIdx.x, t = threadIdx.x;
  h[t] = hg[(size_t)g * DIM + t];
  __syncthreads();
  float acc = 0.f;
#pragma unroll
  for (int jj = 0; jj < 2; jj++) {
    int j = t + jj * 128;
    float s = b1[j];
    for (int k = 0; k < DIM; k++) s += h[k] * W1[k * 256 + j];
    acc += fmaxf(s, 0.f) * W2[j];
  }
  red[t] = acc; __syncthreads();
  for (int off = 64; off > 0; off >>= 1) {
    if (t < off) red[t] += red[t + off];
    __syncthreads();
  }
  if (t == 0) out[g] = red[0] + b2[0];
}

// ---------------------------------------------------------------------------
extern "C" void kernel_launch(void* const* d_in, const int* in_sizes, int n_in,
                              void* d_out, int out_size, void* d_ws, size_t ws_size,
                              hipStream_t stream) {
  const int N = in_sizes[0];       // 50000
  const int E = in_sizes[1];       // 600000
  const int L = 3, G = 512;
  const int DD = DIM * DIM;

  const int*   d_d       = (const int*)  d_in[0];
  const int*   d_src     = (const int*)  d_in[1];
  const int*   d_dst     = (const int*)  d_in[2];
  const int*   d_bid     = (const int*)  d_in[3];
  const float* d_embed   = (const float*)d_in[4];
  const float* d_linW    = (const float*)d_in[5];
  const float* d_linb    = (const float*)d_in[6];
  const float* d_mlpW1   = (const float*)d_in[7];
  const float* d_mlpg1   = (const float*)d_in[8];
  const float* d_mlpb1   = (const float*)d_in[9];
  const float* d_mlpW2   = (const float*)d_in[10];
  const float* d_eps     = (const float*)d_in[11];
  const float* d_bng     = (const float*)d_in[12];
  const float* d_bnb     = (const float*)d_in[13];
  const float* d_poolW1  = (const float*)d_in[14];
  const float* d_poolg1  = (const float*)d_in[15];
  const float* d_poolb1  = (const float*)d_in[16];
  const float* d_poolW2  = (const float*)d_in[17];
  const float* d_pbng    = (const float*)d_in[18];
  const float* d_pbnb    = (const float*)d_in[19];
  const float* d_predW1  = (const float*)d_in[20];
  const float* d_predb1  = (const float*)d_in[21];
  const float* d_predW2  = (const float*)d_in[22];
  const float* d_predb2  = (const float*)d_in[23];

  float* out = (float*)d_out;            // [pred 512 | h_graph G*D | h_means L*G*D]

  // workspace carve-up
  const size_t Nd = (size_t)N * DIM;
  float* bufA   = (float*)d_ws;          // node features x
  float* bufB   = bufA + Nd;             // y / h
  float* bufC   = bufB + Nd;             // z (eps residual + scatter target)
  float* bufD   = bufC + Nd;             // t (pre-BN matmul result)
  float* counts = bufD + Nd;             // [G]
  float* colsum = counts + G;            // [128] (+[128] sumsq contiguous)
  float* colsq  = colsum + DIM;
  float* sc1    = colsq + DIM;           // [128]
  float* sh1    = sc1 + DIM;             // [128]
  float* hsum   = sh1 + DIM;             // [G*D]

  const int TPB = 256;
  const int ndGrid   = (int)((Nd + TPB - 1) / TPB);
  const int gemmGrid = (N + 127) / 128;
  const int edgeGrid = (int)(((size_t)E * 32 + TPB - 1) / TPB);
  const float invN = 1.0f / (float)N;

  // counts (once)
  fill0_kernel<<<(G + TPB - 1) / TPB, TPB, 0, stream>>>(counts, G);
  counts_kernel<<<(N + TPB - 1) / TPB, TPB, 0, stream>>>(d_bid, counts, N);

  // x = embed_d[clip(d)]
  embed_kernel<<<ndGrid, TPB, 0, stream>>>(d_d, d_embed, bufA, N);

  for (int l = 0; l < L; l++) {
    // y = relu(x @ lin_W + lin_b)
    gemm128_kernel<0, 1><<<gemmGrid, TPB, 0, stream>>>(
        bufA, d_linW + (size_t)l * DD, nullptr, nullptr, d_linb + l * DIM, bufB, N);
    // z = x * (1 + eps)
    scale_kernel<<<ndGrid, TPB, 0, stream>>>(bufA, d_eps, l, bufC, (int)Nd);
    // z[dst] += y[src]
    edge_kernel<<<edgeGrid, TPB, 0, stream>>>(d_src, d_dst, bufB, bufC, E);
    // t = z @ mlp_W1
    gemm128_kernel<0, 0><<<gemmGrid, TPB, 0, stream>>>(
        bufC, d_mlpW1 + (size_t)l * DD, nullptr, nullptr, nullptr, bufD, N);
    // BN stats of t -> fused affine
    fill0_kernel<<<1, TPB, 0, stream>>>(colsum, 2 * DIM);
    colstats_kernel<<<256, TPB, 0, stream>>>(bufD, colsum, colsq, N);
    bnfin_kernel<<<1, DIM, 0, stream>>>(colsum, colsq, d_mlpg1 + l * DIM, d_mlpb1 + l * DIM,
                                        invN, sc1, sh1);
    // h = relu(bn(t)) @ mlp_W2
    gemm128_kernel<1, 0><<<gemmGrid, TPB, 0, stream>>>(
        bufD, d_mlpW2 + (size_t)l * DD, sc1, sh1, nullptr, bufB, N);
    // outer BN stats of h
    fill0_kernel<<<1, TPB, 0, stream>>>(colsum, 2 * DIM);
    colstats_kernel<<<256, TPB, 0, stream>>>(bufB, colsum, colsq, N);
    bnfin_kernel<<<1, DIM, 0, stream>>>(colsum, colsq, d_bng + l * DIM, d_bnb + l * DIM,
                                        invN, sc1, sh1);
    // x = relu(bn(h)), accumulate per-graph sums
    fill0_kernel<<<(G * DIM + TPB - 1) / TPB, TPB, 0, stream>>>(hsum, G * DIM);
    bnrelu_kernel<1><<<ndGrid, TPB, 0, stream>>>(bufB, sc1, sh1, bufA, d_bid, hsum, N);
    // h_means[l] = segment mean
    segfin_kernel<<<(G * DIM + TPB - 1) / TPB, TPB, 0, stream>>>(
        hsum, counts, out + 512 + (size_t)G * DIM + (size_t)l * G * DIM, G * DIM);
  }

  // Pool: p = relu(bn(x @ pool_W1)) @ pool_W2 ; p = relu(bn(p)) ; segment mean
  gemm128_kernel<0, 0><<<gemmGrid, TPB, 0, stream>>>(
      bufA, d_poolW1, nullptr, nullptr, nullptr, bufD, N);
  fill0_kernel<<<1, TPB, 0, stream>>>(colsum, 2 * DIM);
  colstats_kernel<<<256, TPB, 0, stream>>>(bufD, colsum, colsq, N);
  bnfin_kernel<<<1, DIM, 0, stream>>>(colsum, colsq, d_poolg1, d_poolb1, invN, sc1, sh1);
  gemm128_kernel<1, 0><<<gemmGrid, TPB, 0, stream>>>(
      bufD, d_poolW2, sc1, sh1, nullptr, bufB, N);
  fill0_kernel<<<1, TPB, 0, stream>>>(colsum, 2 * DIM);
  colstats_kernel<<<256, TPB, 0, stream>>>(bufB, colsum, colsq, N);
  bnfin_kernel<<<1, DIM, 0, stream>>>(colsum, colsq, d_pbng, d_pbnb, invN, sc1, sh1);
  fill0_kernel<<<(G * DIM + TPB - 1) / TPB, TPB, 0, stream>>>(hsum, G * DIM);
  bnrelu_kernel<1><<<ndGrid, TPB, 0, stream>>>(bufB, sc1, sh1, bufC, d_bid, hsum, N);
  segfin_kernel<<<(G * DIM + TPB - 1) / TPB, TPB, 0, stream>>>(
      hsum, counts, out + 512, G * DIM);   // h_graph

  // prediction = relu(h_graph @ pred_W1 + b1) @ pred_W2 + b2
  pred_kernel<<<G, DIM, 0, stream>>>(out + 512, d_predW1, d_predb1, d_predW2, d_predb2, out);
}
